// MoE_net_37941741092910
// MI455X (gfx1250) — matrix operand order repaired and checked by reading, not compiled, not run on hardware
//
#include <hip/hip_runtime.h>

#define B_TOT 65536
#define OBS   256
#define ACT   32
#define NEXP  8
#define H1D   512
#define H2D   256
#define H3D   128
#define GHD   256

typedef __attribute__((ext_vector_type(16))) __bf16 v16bf;
typedef __attribute__((ext_vector_type(8)))  __bf16 v8bf;
typedef __attribute__((ext_vector_type(8)))  float  v8f;

__device__ __forceinline__ v8f vzero8() {
  v8f z = {0.f, 0.f, 0.f, 0.f, 0.f, 0.f, 0.f, 0.f};
  return z;
}

__device__ __forceinline__ float elu1(float v) {
  return v > 0.f ? v : (__expf(v) - 1.f);
}

__device__ __forceinline__ v8f wmma_bf16(v16bf a, v16bf b, v8f c) {
  // D = A(16x32 bf16) * B(32x16 bf16) + C(16x16 f32)
  return __builtin_amdgcn_wmma_f32_16x16x32_bf16(false, a, false, b,
                                                 (short)0, c, false, false);
}

// A-fragment (16x32 bf16) from LDS, row-major storage with padded stride.
// ISA layout: lane<16 holds row M=lane, K={0..7,16..23}+base; lane>=16 K={8..15,24..31}.
__device__ __forceinline__ v16bf lds_frag_a(const __bf16* p) {
  v8bf lo = *(const v8bf*)(p);
  v8bf hi = *(const v8bf*)(p + 16);
  return __builtin_shufflevector(lo, hi, 0, 1, 2, 3, 4, 5, 6, 7,
                                 8, 9, 10, 11, 12, 13, 14, 15);
}

// Build the 8 A-fragments of this wave's 16 x-rows (K=256) directly from global fp32.
__device__ __forceinline__ void load_x_frags(const float* __restrict__ xrow,
                                             v16bf* xa, int khalf) {
#pragma unroll
  for (int kt = 0; kt < 8; ++kt) {
    const int k0 = kt * 32 + khalf;
    v16bf a;
#pragma unroll
    for (int t = 0; t < 8; ++t) {
      a[t]     = (__bf16)xrow[k0 + t];
      a[t + 8] = (__bf16)xrow[k0 + 16 + t];
    }
    xa[kt] = a;
  }
}

// -------------------------------------------------------------------------
// Pack fp32 weight [E][K][N] -> bf16 WMMA-B fragment-major layout:
// idx = ((e*NT + nt)*KT + kt)*512 + lane*16 + j
// element = W[e][kt*32 + (lane>=16?8:0) + (j&7) + (j>=8?16:0)][nt*16 + (lane&15)]  (0 if OOB pad)
// -------------------------------------------------------------------------
__global__ __launch_bounds__(256) void pack_wmma_b(
    const float* __restrict__ src, __bf16* __restrict__ dst,
    int K, int N, int KT, int NT, int total) {
  const int idx = blockIdx.x * blockDim.x + threadIdx.x;
  if (idx >= total) return;
  const int q    = idx & 511;
  const int tile = idx >> 9;
  const int kt   = tile % KT;
  const int nt   = (tile / KT) % NT;
  const int e    = tile / (KT * NT);
  const int ln   = q >> 4;
  const int j    = q & 15;
  const int k    = kt * 32 + ((ln & 16) ? 8 : 0) + (j & 7) + ((j & 8) ? 16 : 0);
  const int col  = nt * 16 + (ln & 15);
  float v = 0.f;
  if (k < K && col < N) v = src[((size_t)e * K + k) * N + col];
  dst[idx] = (__bf16)v;
}

// -------------------------------------------------------------------------
// Gate: logits = elu(x@gW1+gb1)@gW2+gb2 ; top-2 softmax -> gatew[B,8]
// -------------------------------------------------------------------------
__global__ __launch_bounds__(256) void moe_gate(
    const float* __restrict__ x,
    const __bf16* __restrict__ pkG1, const float* __restrict__ gb1,
    const __bf16* __restrict__ pkG2, const float* __restrict__ gb2,
    float* __restrict__ gatew) {
  __shared__ __bf16 sH[32 * 264];   // hidden [32 x 256] bf16, padded stride
  __shared__ float  sLog[32 * 16];  // logits (cols >= 8 unused)

  const int tid  = threadIdx.x;
  const int lane = tid & 31;
  const int wave = tid >> 5;   // 0..7
  const int mrow = wave >> 2;  // 0..1
  const int ngrp = wave & 3;   // 0..3
  const int blockRow = blockIdx.x * 32;

  const int khalf   = (lane & 16) ? 8 : 0;
  const int rowOffC = (lane & 16) ? 8 : 0;
  const int colC    = lane & 15;

  v16bf xa[8];
  load_x_frags(x + (size_t)(blockRow + mrow * 16 + (lane & 15)) * OBS, xa, khalf);

  // L1g: [32x256] @ [256x256] ; each wave: 4 n-tiles
  {
    v8f c[4];
#pragma unroll
    for (int j = 0; j < 4; ++j) c[j] = vzero8();
    for (int kt = 0; kt < 8; ++kt) {
      v16bf a = xa[kt];
#pragma unroll
      for (int j = 0; j < 4; ++j) {
        const int nt = ngrp * 4 + j;
        v16bf b = *(const v16bf*)(pkG1 + ((size_t)nt * 8 + kt) * 512 + lane * 16);
        c[j] = wmma_bf16(a, b, c[j]);
      }
    }
#pragma unroll
    for (int j = 0; j < 4; ++j) {
      const int col  = (ngrp * 4 + j) * 16 + colC;
      const float bv = gb1[col];
#pragma unroll
      for (int i = 0; i < 8; ++i) {
        const int row = mrow * 16 + rowOffC + i;
        sH[row * 264 + col] = (__bf16)elu1(c[j][i] + bv);
      }
    }
  }
  __syncthreads();

  // L2g: [32x256] @ [256x16(pad)] -> logits; 2 waves (ngrp==0) compute
  if (ngrp == 0) {
    v8f c = vzero8();
    for (int kt = 0; kt < 8; ++kt) {
      v16bf a = lds_frag_a(&sH[(mrow * 16 + (lane & 15)) * 264 + kt * 32 + khalf]);
      v16bf b = *(const v16bf*)(pkG2 + (size_t)kt * 512 + lane * 16);
      c = wmma_bf16(a, b, c);
    }
    const float bv = (colC < NEXP) ? gb2[colC] : 0.f;
#pragma unroll
    for (int i = 0; i < 8; ++i) {
      const int row = mrow * 16 + rowOffC + i;
      sLog[row * 16 + colC] = c[i] + bv;
    }
  }
  __syncthreads();

  // Top-2 softmax per row
  if (tid < 32) {
    const int row = tid;
    float lg[NEXP];
#pragma unroll
    for (int e = 0; e < NEXP; ++e) lg[e] = sLog[row * 16 + e];
    int m1 = 0;
#pragma unroll
    for (int e = 1; e < NEXP; ++e)
      if (lg[e] > lg[m1]) m1 = e;
    int m2 = (m1 == 0) ? 1 : 0;
#pragma unroll
    for (int e = 0; e < NEXP; ++e)
      if (e != m1 && lg[e] > lg[m2]) m2 = e;
    const float ebv   = __expf(lg[m2] - lg[m1]);
    const float denom = 1.f + ebv;
    const float w1 = 1.f / denom;
    const float w2 = ebv / denom;
    const size_t rb = (size_t)(blockRow + row) * NEXP;
#pragma unroll
    for (int e = 0; e < NEXP; ++e)
      gatew[rb + e] = (e == m1) ? w1 : ((e == m2) ? w2 : 0.f);
  }
}

// -------------------------------------------------------------------------
// Fused expert MLPs + gate-weighted combine.
// Block: 32 batch rows, 8 waves = 2(M) x 4(N). Intermediates live in LDS bf16.
// -------------------------------------------------------------------------
__global__ __launch_bounds__(256) void moe_experts(
    const float* __restrict__ x,
    const __bf16* __restrict__ pkW1, const float* __restrict__ eb1,
    const __bf16* __restrict__ pkW2, const float* __restrict__ eb2,
    const __bf16* __restrict__ pkW3, const float* __restrict__ eb3,
    const __bf16* __restrict__ pkW4, const float* __restrict__ eb4,
    const float* __restrict__ gatew,
    float* __restrict__ out) {
  __shared__ __bf16 sA[32 * 520];  // h1 [32x512] (stride 520); reused as h3 [32x128] (stride 136)
  __shared__ __bf16 sB[32 * 264];  // h2 [32x256] (stride 264)
  __shared__ float  sG[32 * NEXP];

  const int tid  = threadIdx.x;
  const int lane = tid & 31;
  const int wave = tid >> 5;
  const int mrow = wave >> 2;  // 0..1
  const int ngrp = wave & 3;   // 0..3
  const int blockRow = blockIdx.x * 32;

  const int khalf   = (lane & 16) ? 8 : 0;
  const int rowOffC = (lane & 16) ? 8 : 0;
  const int colC    = lane & 15;

  sG[tid] = gatew[(size_t)blockRow * NEXP + tid];  // 32*8 == 256

  // x A-fragments stay in registers across all experts / the whole K loop of L1
  v16bf xa[8];
  load_x_frags(x + (size_t)(blockRow + mrow * 16 + (lane & 15)) * OBS, xa, khalf);
  __syncthreads();

  v8f outAcc = vzero8();

  for (int e = 0; e < NEXP; ++e) {
    // ---- L1: x[32x256] @ W1[256x512] -> h1 (sA) ----
    {
      v8f c[8];
#pragma unroll
      for (int j = 0; j < 8; ++j) c[j] = vzero8();
      const __bf16* wb = pkW1 + (size_t)e * (32 * 8) * 512;
      for (int kt = 0; kt < 8; ++kt) {
        v16bf a = xa[kt];
#pragma unroll
        for (int j = 0; j < 8; ++j) {
          const int nt = ngrp * 8 + j;
          v16bf b = *(const v16bf*)(wb + ((size_t)nt * 8 + kt) * 512 + lane * 16);
          c[j] = wmma_bf16(a, b, c[j]);
        }
      }
#pragma unroll
      for (int j = 0; j < 8; ++j) {
        const int col  = (ngrp * 8 + j) * 16 + colC;
        const float bv = eb1[e * H1D + col];
#pragma unroll
        for (int i = 0; i < 8; ++i) {
          const int row = mrow * 16 + rowOffC + i;
          sA[row * 520 + col] = (__bf16)elu1(c[j][i] + bv);
        }
      }
    }
    __syncthreads();

    // ---- L2: h1[32x512] @ W2[512x256] -> h2 (sB) ----
    {
      v8f c[4];
#pragma unroll
      for (int j = 0; j < 4; ++j) c[j] = vzero8();
      const __bf16* wb = pkW2 + (size_t)e * (16 * 16) * 512;
      for (int kt = 0; kt < 16; ++kt) {
        v16bf a = lds_frag_a(&sA[(mrow * 16 + (lane & 15)) * 520 + kt * 32 + khalf]);
#pragma unroll
        for (int j = 0; j < 4; ++j) {
          const int nt = ngrp * 4 + j;
          v16bf b = *(const v16bf*)(wb + ((size_t)nt * 16 + kt) * 512 + lane * 16);
          c[j] = wmma_bf16(a, b, c[j]);
        }
      }
#pragma unroll
      for (int j = 0; j < 4; ++j) {
        const int col  = (ngrp * 4 + j) * 16 + colC;
        const float bv = eb2[e * H2D + col];
#pragma unroll
        for (int i = 0; i < 8; ++i) {
          const int row = mrow * 16 + rowOffC + i;
          sB[row * 264 + col] = (__bf16)elu1(c[j][i] + bv);
        }
      }
    }
    __syncthreads();

    // ---- L3: h2[32x256] @ W3[256x128] -> h3 (sA, stride 136; h1 is dead) ----
    {
      v8f c[2];
      c[0] = vzero8();
      c[1] = vzero8();
      const __bf16* wb = pkW3 + (size_t)e * (8 * 8) * 512;
      for (int kt = 0; kt < 8; ++kt) {
        v16bf a = lds_frag_a(&sB[(mrow * 16 + (lane & 15)) * 264 + kt * 32 + khalf]);
#pragma unroll
        for (int j = 0; j < 2; ++j) {
          const int nt = ngrp * 2 + j;
          v16bf b = *(const v16bf*)(wb + ((size_t)nt * 8 + kt) * 512 + lane * 16);
          c[j] = wmma_bf16(a, b, c[j]);
        }
      }
#pragma unroll
      for (int j = 0; j < 2; ++j) {
        const int col  = (ngrp * 2 + j) * 16 + colC;
        const float bv = eb3[e * H3D + col];
#pragma unroll
        for (int i = 0; i < 8; ++i) {
          const int row = mrow * 16 + rowOffC + i;
          sA[row * 136 + col] = (__bf16)elu1(c[j][i] + bv);
        }
      }
    }
    __syncthreads();

    // ---- L4: h3[32x128] @ W4[128x32] + gate-weighted accumulate ----
    if (ngrp < 2) {  // wave-uniform branch: EXEC stays all-ones for WMMA
      v8f c = vzero8();
      const __bf16* wb = pkW4 + (size_t)e * (2 * 4) * 512;
#pragma unroll
      for (int kt = 0; kt < 4; ++kt) {
        v16bf a = lds_frag_a(&sA[(mrow * 16 + (lane & 15)) * 136 + kt * 32 + khalf]);
        v16bf b = *(const v16bf*)(wb + ((size_t)ngrp * 4 + kt) * 512 + lane * 16);
        c = wmma_bf16(a, b, c);
      }
      const int col  = ngrp * 16 + colC;
      const float bv = eb4[e * ACT + col];
#pragma unroll
      for (int i = 0; i < 8; ++i) {
        const int row = mrow * 16 + rowOffC + i;
        outAcc[i] += sG[row * NEXP + e] * (c[i] + bv);
      }
    }
    __syncthreads();  // h3 reads done before next expert overwrites sA
  }

  if (ngrp < 2) {
    const int col = ngrp * 16 + colC;
#pragma unroll
    for (int i = 0; i < 8; ++i) {
      const int row = blockRow + mrow * 16 + rowOffC + i;
      out[(size_t)row * ACT + col] = outAcc[i];
    }
  }
}

// -------------------------------------------------------------------------
// Workspace layout (bytes, all 256-aligned), total ~7.02 MB:
//   pkW1 bf16  : [0,        2097152)
//   pkW2 bf16  : [2097152,  4194304)
//   pkW3 bf16  : [4194304,  4718592)
//   pkW4 bf16  : [4718592,  4784128)
//   pkG1 bf16  : [4784128,  4915200)
//   pkG2 bf16  : [4915200,  4923392)
//   gatew f32  : [4923392,  7020544)
// -------------------------------------------------------------------------
extern "C" void kernel_launch(void* const* d_in, const int* in_sizes, int n_in,
                              void* d_out, int out_size, void* d_ws, size_t ws_size,
                              hipStream_t stream) {
  (void)in_sizes; (void)n_in; (void)out_size; (void)ws_size;
  const float* x   = (const float*)d_in[0];
  const float* eW1 = (const float*)d_in[1];
  const float* eb1 = (const float*)d_in[2];
  const float* eW2 = (const float*)d_in[3];
  const float* eb2 = (const float*)d_in[4];
  const float* eW3 = (const float*)d_in[5];
  const float* eb3 = (const float*)d_in[6];
  const float* eW4 = (const float*)d_in[7];
  const float* eb4 = (const float*)d_in[8];
  const float* gW1 = (const float*)d_in[9];
  const float* gb1 = (const float*)d_in[10];
  const float* gW2 = (const float*)d_in[11];
  const float* gb2 = (const float*)d_in[12];
  float* out = (float*)d_out;

  char* ws = (char*)d_ws;
  __bf16* pkW1 = (__bf16*)(ws + 0);
  __bf16* pkW2 = (__bf16*)(ws + 2097152);
  __bf16* pkW3 = (__bf16*)(ws + 4194304);
  __bf16* pkW4 = (__bf16*)(ws + 4718592);
  __bf16* pkG1 = (__bf16*)(ws + 4784128);
  __bf16* pkG2 = (__bf16*)(ws + 4915200);
  float*  gw   = (float*)(ws + 4923392);

  // Weight repack (fp32 -> bf16 WMMA fragment layout); ~9 MB read, cheap.
  pack_wmma_b<<<4096, 256, 0, stream>>>(eW1, pkW1, 256, 512, 8, 32, 1048576);
  pack_wmma_b<<<4096, 256, 0, stream>>>(eW2, pkW2, 512, 256, 16, 16, 1048576);
  pack_wmma_b<<<1024, 256, 0, stream>>>(eW3, pkW3, 256, 128, 8, 8, 262144);
  pack_wmma_b<<<128,  256, 0, stream>>>(eW4, pkW4, 128, 32, 4, 2, 32768);
  pack_wmma_b<<<256,  256, 0, stream>>>(gW1, pkG1, 256, 256, 8, 16, 65536);
  pack_wmma_b<<<16,   256, 0, stream>>>(gW2, pkG2, 256, 8, 8, 1, 4096);

  moe_gate<<<B_TOT / 32, 256, 0, stream>>>(x, pkG1, gb1, pkG2, gb2, gw);
  moe_experts<<<B_TOT / 32, 256, 0, stream>>>(x, pkW1, eb1, pkW2, eb2, pkW3, eb3,
                                              pkW4, eb4, gw, out);
}
